// SkipableGAT_20289425506940
// MI455X (gfx1250) — matrix operand order repaired and checked by reading, not compiled, not run on hardware
//
#include <hip/hip_runtime.h>
#include <hip/hip_bf16.h>
#include <cstdint>
#include <cstddef>

typedef __attribute__((ext_vector_type(16))) __bf16 v16bf;
typedef __attribute__((ext_vector_type(8)))  float  v8f;

#define NHEAD   8
#define DIMH    16
#define AA      32
#define JJ      19
#define CC      256
#define GD      128
#define QKVW    768        /* 6*GD */
#define BT_TOT  8192       /* 32*256 */
#define M_TOT   (BT_TOT*JJ)
#define NCHUNK  8
#define BT_CH   (BT_TOT/NCHUNK)
#define M_CH    (BT_CH*JJ)
#define NEDGE   138

/* ---- graph: a = adj + adj@adj (17-node tree), diag zeroed -> 70 edges,
   then base->g1(17), g1->base, base->g2(18), g2->base ---- */
__constant__ int SRC_C[NEDGE] = {
  0,0,0,0,0,0, 1,1,1,1,1, 2,2,2, 3,3, 4,4,4,4,4, 5,5,5, 6,6,
  7,7,7,7,7,7,7, 8,8,8,8,8,8,8,8, 9,9,9,9,9, 10,10,
  11,11,11,11,11,11, 12,12,12, 13,13, 14,14,14,14,14,14, 15,15,15, 16,16,
  0,1,2,3,4,5,6,7,8,9,10,11,12,13,14,15,16,
  17,17,17,17,17,17,17,17,17,17,17,17,17,17,17,17,17,
  0,1,2,3,4,5,6,7,8,9,10,11,12,13,14,15,16,
  18,18,18,18,18,18,18,18,18,18,18,18,18,18,18,18,18 };
__constant__ int DST_C[NEDGE] = {
  1,2,4,5,7,8, 0,2,3,4,7, 0,1,3, 1,2, 0,1,5,6,7, 0,4,6, 4,5,
  0,1,4,8,9,11,14, 0,7,9,10,11,12,14,15, 7,8,10,11,14, 8,9,
  7,8,9,12,13,14, 8,11,13, 11,12, 7,8,9,11,15,16, 8,14,16, 14,15,
  17,17,17,17,17,17,17,17,17,17,17,17,17,17,17,17,17,
  0,1,2,3,4,5,6,7,8,9,10,11,12,13,14,15,16,
  18,18,18,18,18,18,18,18,18,18,18,18,18,18,18,18,18,
  0,1,2,3,4,5,6,7,8,9,10,11,12,13,14,15,16 };
/* CSR over SRC for deterministic gathers */
__constant__ int EOFF_C[20] = {0,8,15,20,24,31,36,40,49,59,66,70,78,83,87,95,100,104,121,138};
__constant__ int EIDX_C[NEDGE] = {
  0,1,2,3,4,5,70,104,  6,7,8,9,10,71,105,  11,12,13,72,106,  14,15,73,107,
  16,17,18,19,20,74,108,  21,22,23,75,109,  24,25,76,110,
  26,27,28,29,30,31,32,77,111,  33,34,35,36,37,38,39,40,78,112,
  41,42,43,44,45,79,113,  46,47,80,114,  48,49,50,51,52,53,81,115,
  54,55,56,82,116,  57,58,83,117,  59,60,61,62,63,64,84,118,
  65,66,67,85,119,  68,69,86,120,
  87,88,89,90,91,92,93,94,95,96,97,98,99,100,101,102,103,
  121,122,123,124,125,126,127,128,129,130,131,132,133,134,135,136,137 };

/* ================= tiny zero-fill (zero row for the concat GEMM) ================= */
__global__ __launch_bounds__(256) void zerofill_kernel(float* __restrict__ p, int n) {
  int g = blockIdx.x * 256 + threadIdx.x;
  if (g < n) p[g] = 0.0f;
}

/* ================= weight convert+transpose fp32(K,N) -> bf16(N,K) ================= */
__global__ __launch_bounds__(256) void wconv_kernel(const float* __restrict__ src,
                                                    __bf16* __restrict__ dst,
                                                    int K, int N) {
  int g = blockIdx.x * 256 + threadIdx.x;
  if (g >= K * N) return;
  int n = g / K, k = g - n * K;
  dst[g] = (__bf16)src[(size_t)k * N + n];
}

/* ================= mean over the 17 body nodes ================= */
__global__ __launch_bounds__(256) void mean_kernel(const float* __restrict__ x,
                                                   float* __restrict__ xm) {
  int g = blockIdx.x * 256 + threadIdx.x;     /* 8192*256 threads */
  int bt = g >> 8, c = g & 255;
  const float* p = x + (size_t)bt * 17 * CC + c;
  float s = 0.f;
#pragma unroll
  for (int j = 0; j < 17; ++j) s += p[(size_t)j * CC];
  xm[g] = s * (1.0f / 17.0f);
}

/* ================= bf16 WMMA fragments ================= */
/* A 16x32 layout (contiguous, matches B): lane half kh covers K run [16kh..16kh+15] */
__device__ __forceinline__ v16bf load_a_frag(const float* __restrict__ rowp, int kh) {
  const float4* p0 = (const float4*)(rowp + 16 * kh);
  const float4* p1 = (const float4*)(rowp + 16 * kh + 8);
  float4 u0 = p0[0], u1 = p0[1], w0 = p1[0], w1 = p1[1];
  v16bf f;
  f[0]=(__bf16)u0.x; f[1]=(__bf16)u0.y; f[2]=(__bf16)u0.z; f[3]=(__bf16)u0.w;
  f[4]=(__bf16)u1.x; f[5]=(__bf16)u1.y; f[6]=(__bf16)u1.z; f[7]=(__bf16)u1.w;
  f[8]=(__bf16)w0.x; f[9]=(__bf16)w0.y; f[10]=(__bf16)w0.z; f[11]=(__bf16)w0.w;
  f[12]=(__bf16)w1.x; f[13]=(__bf16)w1.y; f[14]=(__bf16)w1.z; f[15]=(__bf16)w1.w;
  return f;
}

__device__ __forceinline__ v8f wmma_bf16(v16bf a, v16bf b, v8f c) {
  return __builtin_amdgcn_wmma_f32_16x16x32_bf16(false, a, false, b, (short)0, c, false, false);
}

template<int MODE>  /* 0 = plain store, 2 = drop j>=17 rows into d_out */
__device__ __forceinline__ void store_tile(const v8f& c, int mt, int col, int kh, float bv,
                                           float* __restrict__ C, int N) {
#pragma unroll
  for (int r = 0; r < 8; ++r) {
    int row = mt + r + 8 * kh;
    float val = c[r] + bv;
    if constexpr (MODE == 2) {
      int bt = row / JJ, j = row - bt * JJ;
      if (j < 17) C[((size_t)bt * 17 + j) * CC + col] = val;
    } else {
      C[(size_t)row * N + col] = val;
    }
  }
}

/* ================= GEMM: C[M,N] = Arows(fp32,K) @ Wt^T(bf16 N-major) + bias =========
   Block: 256 thr = 8 waves (4M x 2N); wave tile 32x64 = 2x4 WMMA tiles.
   Block tile 128(M) x 128(N).  MODE 0: A plain.  MODE 1: rows from concat([x,0,xm]),
   zero rows resolved to the pre-zeroed scratch row passed in A.  MODE 2: sliced store. */
template<int MODE>
__global__ __launch_bounds__(256) void gemm_bf16(const float* __restrict__ A,
                                                 const float* __restrict__ X,
                                                 const float* __restrict__ XM,
                                                 const __bf16* __restrict__ Wt,
                                                 const float* __restrict__ bias,
                                                 float* __restrict__ C,
                                                 int N, int K) {
  const int lane = threadIdx.x & 31;
  const int wave = threadIdx.x >> 5;
  const int l16  = lane & 15;
  const int kh   = lane >> 4;
  const int mbase = blockIdx.x * 128 + (wave >> 1) * 32;
  const int n0    = blockIdx.y * 128 + (wave & 1) * 64;
  const int m0 = mbase + l16, m1 = m0 + 16;

  const float *pa0, *pa1;
  if constexpr (MODE == 1) {
    int bt0 = m0 / JJ, j0 = m0 - bt0 * JJ;
    pa0 = (j0 < 17) ? (X + ((size_t)bt0 * 17 + j0) * CC)
                    : ((j0 == 18) ? (XM + (size_t)bt0 * CC) : A /* zero row */);
    int bt1 = m1 / JJ, j1 = m1 - bt1 * JJ;
    pa1 = (j1 < 17) ? (X + ((size_t)bt1 * 17 + j1) * CC)
                    : ((j1 == 18) ? (XM + (size_t)bt1 * CC) : A /* zero row */);
  } else {
    pa0 = A + (size_t)m0 * K;
    pa1 = A + (size_t)m1 * K;
  }
  const __bf16* pb = Wt + (size_t)(n0 + l16) * K + 16 * kh;
  const size_t bstride = (size_t)16 * K;   /* next 16-col tile of Wt */

  v8f acc[2][4];
#pragma unroll
  for (int m = 0; m < 2; ++m)
#pragma unroll
    for (int j = 0; j < 4; ++j)
      acc[m][j] = (v8f){0.f,0.f,0.f,0.f,0.f,0.f,0.f,0.f};

  for (int kk = 0; kk < K; kk += 32) {
    __builtin_prefetch(pa0 + 64, 0, 0);
    __builtin_prefetch(pa1 + 64, 0, 0);
    v16bf b0 = *(const v16bf*)(pb);
    v16bf b1 = *(const v16bf*)(pb + bstride);
    v16bf b2 = *(const v16bf*)(pb + 2 * bstride);
    v16bf b3 = *(const v16bf*)(pb + 3 * bstride);
    v16bf a0 = load_a_frag(pa0, kh);
    v16bf a1 = load_a_frag(pa1, kh);
    acc[0][0] = wmma_bf16(a0, b0, acc[0][0]);
    acc[0][1] = wmma_bf16(a0, b1, acc[0][1]);
    acc[0][2] = wmma_bf16(a0, b2, acc[0][2]);
    acc[0][3] = wmma_bf16(a0, b3, acc[0][3]);
    acc[1][0] = wmma_bf16(a1, b0, acc[1][0]);
    acc[1][1] = wmma_bf16(a1, b1, acc[1][1]);
    acc[1][2] = wmma_bf16(a1, b2, acc[1][2]);
    acc[1][3] = wmma_bf16(a1, b3, acc[1][3]);
    pa0 += 32; pa1 += 32; pb += 32;
  }

#pragma unroll
  for (int j = 0; j < 4; ++j) {
    int col = n0 + 16 * j + l16;
    float bv = bias ? bias[col] : 0.0f;
    store_tile<MODE>(acc[0][j], mbase,      col, kh, bv, C, N);
    store_tile<MODE>(acc[1][j], mbase + 16, col, kh, bv, C, N);
  }
}

/* ================= GAT message passing: one block per (b,t) ================= */
__global__ __launch_bounds__(256) void gat_kernel(const float* __restrict__ qkv,
                                                  const float* __restrict__ aw,
                                                  float* __restrict__ out) {
  __shared__ float s_sc[NEDGE * NHEAD];
  __shared__ float s_sig[JJ * NHEAD];
  __shared__ float s_max[NHEAD];
  __shared__ float s_aw[AA];
  const float* q = qkv + (size_t)blockIdx.x * JJ * QKVW;
  const int tid = threadIdx.x;
  if (tid < AA) s_aw[tid] = aw[tid];
  __syncthreads();

  /* edge scores: s = sum_a softplus(q[src]+k[dst]) * aw */
  for (int p = tid; p < NEDGE * NHEAD; p += 256) {
    int e = p >> 3, h = p & 7;
    const float* qp = q + SRC_C[e] * QKVW + h * 96;
    const float* kp = q + DST_C[e] * QKVW + h * 96 + 32;
    float acc = 0.f;
#pragma unroll 4
    for (int a = 0; a < AA; ++a) {
      float xv = qp[a] + kp[a];
      float z = fmaxf(xv, 0.f) + log1pf(expf(-fabsf(xv)));   /* stable softplus */
      acc = fmaf(z, s_aw[a], acc);
    }
    s_sc[p] = acc;
  }
  __syncthreads();

  /* global (over edges) max per head */
  if (tid < NHEAD) {
    float m = -3.0e38f;
    for (int e = 0; e < NEDGE; ++e) m = fmaxf(m, s_sc[e * 8 + tid]);
    s_max[tid] = m;
  }
  __syncthreads();

  for (int p = tid; p < NEDGE * NHEAD; p += 256)
    s_sc[p] = expf(s_sc[p] - s_max[p & 7]);
  __syncthreads();

  /* sigma per (node, head) — deterministic gather over CSR */
  for (int i = tid; i < JJ * NHEAD; i += 256) {
    int n = i >> 3, h = i & 7;
    float s = 1e-10f;
    for (int t = EOFF_C[n]; t < EOFF_C[n + 1]; ++t) s += s_sc[EIDX_C[t] * 8 + h];
    s_sig[i] = s;
  }
  __syncthreads();

  /* out = v1 + (sum_e s_e * v2[dst_e]) / sigma */
  float* o = out + (size_t)blockIdx.x * JJ * GD;
  for (int i = tid; i < JJ * GD; i += 256) {
    int n = i >> 7, c = i & 127, h = c >> 4, d = c & 15;
    float acc = 0.f;
    for (int t = EOFF_C[n]; t < EOFF_C[n + 1]; ++t) {
      int e = EIDX_C[t];
      acc = fmaf(s_sc[e * 8 + h], q[DST_C[e] * QKVW + h * 96 + 80 + d], acc);
    }
    float v1 = q[n * QKVW + h * 96 + 64 + d];
    o[i] = v1 + acc / s_sig[n * 8 + h];
  }
}

/* ================= LayerNorm (rows of 256, one wave per row) ================= */
__global__ __launch_bounds__(256) void ln_kernel(float* __restrict__ h,
                                                 const float* __restrict__ g,
                                                 const float* __restrict__ b) {
  const int lane = threadIdx.x & 31;
  const int wave = threadIdx.x >> 5;
  float* p = h + ((size_t)blockIdx.x * 8 + wave) * CC;
  float4 u = *(const float4*)(p + lane * 4);
  float4 v = *(const float4*)(p + 128 + lane * 4);
  float s = u.x + u.y + u.z + u.w + v.x + v.y + v.z + v.w;
  float q = u.x*u.x + u.y*u.y + u.z*u.z + u.w*u.w
          + v.x*v.x + v.y*v.y + v.z*v.z + v.w*v.w;
#pragma unroll
  for (int m = 16; m > 0; m >>= 1) { s += __shfl_xor(s, m, 32); q += __shfl_xor(q, m, 32); }
  float mean = s * (1.0f / 256.0f);
  float var  = q * (1.0f / 256.0f) - mean * mean;
  float rstd = rsqrtf(var + 1e-5f);
  int c0 = lane * 4, c1 = 128 + lane * 4;
  float4 og, oh;
  og.x = (u.x - mean) * rstd * g[c0 + 0] + b[c0 + 0];
  og.y = (u.y - mean) * rstd * g[c0 + 1] + b[c0 + 1];
  og.z = (u.z - mean) * rstd * g[c0 + 2] + b[c0 + 2];
  og.w = (u.w - mean) * rstd * g[c0 + 3] + b[c0 + 3];
  oh.x = (v.x - mean) * rstd * g[c1 + 0] + b[c1 + 0];
  oh.y = (v.y - mean) * rstd * g[c1 + 1] + b[c1 + 1];
  oh.z = (v.z - mean) * rstd * g[c1 + 2] + b[c1 + 2];
  oh.w = (v.w - mean) * rstd * g[c1 + 3] + b[c1 + 3];
  *(float4*)(p + lane * 4)       = og;
  *(float4*)(p + 128 + lane * 4) = oh;
}

/* ================= driver ================= */
extern "C" void kernel_launch(void* const* d_in, const int* in_sizes, int n_in,
                              void* d_out, int out_size, void* d_ws, size_t ws_size,
                              hipStream_t stream) {
  const float* x      = (const float*)d_in[0];
  const float* w1_in  = (const float*)d_in[1];
  const float* b1_in  = (const float*)d_in[2];
  const float* wqkv1  = (const float*)d_in[3];
  const float* a1     = (const float*)d_in[4];
  const float* w1_out = (const float*)d_in[5];
  const float* b1_out = (const float*)d_in[6];
  const float* ln_g   = (const float*)d_in[7];
  const float* ln_b   = (const float*)d_in[8];
  const float* w2_in  = (const float*)d_in[9];
  const float* b2_in  = (const float*)d_in[10];
  const float* wqkv2  = (const float*)d_in[11];
  const float* a2     = (const float*)d_in[12];
  const float* w2_out = (const float*)d_in[13];
  const float* b2_out = (const float*)d_in[14];
  float* out = (float*)d_out;

  char* ws = (char*)d_ws;
  const size_t OFF_XM  = 0;                  /*   8,388,608 */
  const size_t OFF_H   = 8388608ull;         /* 159,383,552 */
  const size_t OFF_ACT = 167772160ull;       /*  79,691,776 */
  const size_t OFF_QKV = 247463936ull;       /*  59,768,832 */
  const size_t OFF_WT  = 307232768ull;       /* 6 x 512KB   */
  const size_t OFF_Z   = 310378496ull;       /* 1KB zero row */
  float*  xm  = (float*)(ws + OFF_XM);
  float*  hb  = (float*)(ws + OFF_H);
  float*  act = (float*)(ws + OFF_ACT);
  float*  qkv = (float*)(ws + OFF_QKV);
  float*  zrow = (float*)(ws + OFF_Z);
  __bf16* w1in_t  = (__bf16*)(ws + OFF_WT + 0ull * 524288);
  __bf16* wqkv1_t = (__bf16*)(ws + OFF_WT + 1ull * 524288);
  __bf16* w1out_t = (__bf16*)(ws + OFF_WT + 2ull * 524288);
  __bf16* w2in_t  = (__bf16*)(ws + OFF_WT + 3ull * 524288);
  __bf16* wqkv2_t = (__bf16*)(ws + OFF_WT + 4ull * 524288);
  __bf16* w2out_t = (__bf16*)(ws + OFF_WT + 5ull * 524288);

  /* init: zero row + weights -> bf16 transposed (N,K) */
  zerofill_kernel<<<1, 256, 0, stream>>>(zrow, 256);
  wconv_kernel<<<(CC*GD + 255) / 256,   256, 0, stream>>>(w1_in,  w1in_t,  CC, GD);
  wconv_kernel<<<(GD*QKVW + 255) / 256, 256, 0, stream>>>(wqkv1,  wqkv1_t, GD, QKVW);
  wconv_kernel<<<(GD*CC + 255) / 256,   256, 0, stream>>>(w1_out, w1out_t, GD, CC);
  wconv_kernel<<<(CC*GD + 255) / 256,   256, 0, stream>>>(w2_in,  w2in_t,  CC, GD);
  wconv_kernel<<<(GD*QKVW + 255) / 256, 256, 0, stream>>>(wqkv2,  wqkv2_t, GD, QKVW);
  wconv_kernel<<<(GD*CC + 255) / 256,   256, 0, stream>>>(w2_out, w2out_t, GD, CC);

  /* xm = mean over 17 body nodes */
  mean_kernel<<<BT_TOT, 256, 0, stream>>>(x, xm);

  /* ---------- block 1 ---------- */
  gemm_bf16<1><<<dim3(M_TOT/128, GD/128), 256, 0, stream>>>(zrow, x, xm, w1in_t, b1_in, act, GD, CC);
  for (int c = 0; c < NCHUNK; ++c) {
    float* actc = act + (size_t)c * M_CH * GD;
    gemm_bf16<0><<<dim3(M_CH/128, QKVW/128), 256, 0, stream>>>(actc, nullptr, nullptr, wqkv1_t, nullptr, qkv, QKVW, GD);
    gat_kernel<<<BT_CH, 256, 0, stream>>>(qkv, a1, actc);
  }
  gemm_bf16<0><<<dim3(M_TOT/128, CC/128), 256, 0, stream>>>(act, nullptr, nullptr, w1out_t, b1_out, hb, CC, GD);
  ln_kernel<<<M_TOT / 8, 256, 0, stream>>>(hb, ln_g, ln_b);

  /* ---------- block 2 ---------- */
  gemm_bf16<0><<<dim3(M_TOT/128, GD/128), 256, 0, stream>>>(hb, nullptr, nullptr, w2in_t, b2_in, act, GD, CC);
  for (int c = 0; c < NCHUNK; ++c) {
    float* actc = act + (size_t)c * M_CH * GD;
    gemm_bf16<0><<<dim3(M_CH/128, QKVW/128), 256, 0, stream>>>(actc, nullptr, nullptr, wqkv2_t, nullptr, qkv, QKVW, GD);
    gat_kernel<<<BT_CH, 256, 0, stream>>>(qkv, a2, actc);
  }
  gemm_bf16<2><<<dim3(M_TOT/128, CC/128), 256, 0, stream>>>(act, nullptr, nullptr, w2out_t, b2_out, out, CC, GD);

  (void)in_sizes; (void)n_in; (void)out_size; (void)ws_size;
}
